// OpenPanguMoE_16020228014081
// MI455X (gfx1250) — compile-verified, matrix operand order checked by reading, not run on hardware
//
#include <hip/hip_runtime.h>
#include <hip/hip_bf16.h>
#include <math.h>

// Problem sizes (fixed by the reference)
#define T_TOK 1024
#define H_DIM 1024
#define I_DIM 512
#define E_NUM 16
#define TOPK  4
#define SI    512                      // SHARED_I
#define TOTI  (E_NUM * I_DIM + SI)     // 8704: concatenated intermediate dim

typedef __bf16 bf16;
typedef __attribute__((ext_vector_type(16))) __bf16 v16bf;
typedef __attribute__((ext_vector_type(8)))  float  v8f;

union V16U { v16bf v; uint4 q[2]; };

__device__ __forceinline__ unsigned short f2bf(float f) {
  unsigned int u = __float_as_uint(f);
  u += 0x7fffu + ((u >> 16) & 1u);     // round-to-nearest-even
  return (unsigned short)(u >> 16);
}

__device__ __forceinline__ void load_frag(V16U& d, const bf16* p) {
  d.q[0] = *(const uint4*)(p);
  d.q[1] = *(const uint4*)(p + 16);
}

// ---------------- conversion / transpose kernels (fp32 -> bf16) ----------------

__global__ void k_cvt_x(const float* __restrict__ x, unsigned short* __restrict__ o) {
  int i = blockIdx.x * 256 + threadIdx.x;          // T*H = 1M
  o[i] = f2bf(x[i]);
}

// w_gate_up [E][H][2I] -> wguT [E][2I][H]   (N-major so B loads are contiguous in K)
__global__ void k_cvt_wguT(const float* __restrict__ w, unsigned short* __restrict__ o) {
  unsigned i = blockIdx.x * 256u + threadIdx.x;    // E*2I*H = 16M
  unsigned e = i >> 20;                            // / (2I*H) = 2^20
  unsigned r = i & 1048575u;
  unsigned n = r >> 10;                            // / H
  unsigned k = r & 1023u;
  o[i] = f2bf(w[(e << 20) + (k << 10) + n]);       // in[e][k][n], H*2I = 2^20
}

// shared_gate_up [H][2*SI] -> sguT [2*SI][H]
__global__ void k_cvt_sguT(const float* __restrict__ w, unsigned short* __restrict__ o) {
  unsigned i = blockIdx.x * 256u + threadIdx.x;    // 2*SI*H = 1M
  unsigned n = i >> 10;
  unsigned k = i & 1023u;
  o[i] = f2bf(w[(k << 10) + n]);
}

// merged down weights, transposed: wdT [H][TOTI]
//  kk < 8192 : w_down[e][ii][n]  (e = kk/512, ii = kk%512)
//  kk >= 8192: shared_down[kk-8192][n]
__global__ void k_cvt_wdT(const float* __restrict__ wd, const float* __restrict__ sd,
                          unsigned short* __restrict__ o) {
  unsigned i = blockIdx.x * 256u + threadIdx.x;    // H*TOTI = 8,912,896
  unsigned n  = i / TOTI;
  unsigned kk = i % TOTI;
  float v;
  if (kk < (unsigned)(E_NUM * I_DIM)) {
    unsigned e = kk >> 9, ii = kk & 511u;
    v = wd[e * (unsigned)(I_DIM * H_DIM) + ii * (unsigned)H_DIM + n];
  } else {
    unsigned ii = kk - (unsigned)(E_NUM * I_DIM);
    v = sd[ii * (unsigned)H_DIM + n];
  }
  o[i] = f2bf(v);
}

// ---------------- router: fp32 logits -> sigmoid -> top-4 -> dense route[T][E] ----------------

__global__ void k_router(const float* __restrict__ x, const float* __restrict__ gw,
                         float* __restrict__ route) {
  __shared__ float part[128];
  __shared__ float sc[E_NUM];
  const int t   = blockIdx.x;
  const int tid = threadIdx.x;                     // 128 threads
  const int e   = tid & 15;
  const int seg = tid >> 4;                        // 8 segments of 128 H-elems
  const float* xr = x  + (size_t)t * H_DIM;
  const float* gr = gw + (size_t)e * H_DIM;
  float p = 0.f;
  const int base = seg * 128;
  for (int h = 0; h < 128; ++h) p += xr[base + h] * gr[base + h];
  part[tid] = p;
  __syncthreads();
  if (tid < E_NUM) {
    float s = 0.f;
    for (int g = 0; g < 8; ++g) s += part[(g << 4) + tid];
    sc[tid] = 1.f / (1.f + __expf(-s));            // sigmoid
  }
  __syncthreads();
  if (tid == 0) {
    float w[E_NUM]; bool used[E_NUM];
    for (int i = 0; i < E_NUM; ++i) { w[i] = 0.f; used[i] = false; }
    int   sel[TOPK]; float sv[TOPK]; float sum = 0.f;
    for (int k = 0; k < TOPK; ++k) {               // ties -> lowest index (matches top_k)
      float best = -1.f; int bi = 0;
      for (int i = 0; i < E_NUM; ++i)
        if (!used[i] && sc[i] > best) { best = sc[i]; bi = i; }
      used[bi] = true; sel[k] = bi; sv[k] = best; sum += best;
    }
    const float inv = 1.f / sum;
    for (int k = 0; k < TOPK; ++k) w[sel[k]] = sv[k] * inv;
    for (int i = 0; i < E_NUM; ++i) route[t * E_NUM + i] = w[i];
  }
}

// ---------------- fused up-projection + SwiGLU + route scaling ----------------
// One wave owns a 64-row x 16-col strip of hbuf: 4 M-tiles x (gate,up) pair
// = 8 accumulators; A fragments reused across gate/up, B fragments reused
// across the four row tiles -> 8 WMMAs per 6 fragment loads (~21 FLOP/B).
// 16 M-superblocks x 544 N-tiles = 8704 waves; 8 waves/block -> 1088 blocks.
__global__ void k_up_swiglu(const bf16* __restrict__ xb,
                            const bf16* __restrict__ wguT,
                            const bf16* __restrict__ sguT,
                            const float* __restrict__ route,
                            unsigned short* __restrict__ hbuf) {
  const int wave = threadIdx.x >> 5;
  const int lane = threadIdx.x & 31;
  const int half = lane >> 4;
  const int l16  = lane & 15;
  const int tile   = blockIdx.x * (blockDim.x >> 5) + wave;
  const int tile_n = tile % 544;
  const int tile_m = tile / 544;                   // 0..15
  const int row0   = tile_m << 6;                  // 64 rows per wave

  const bf16 *bg, *bu;
  int e = -1, col0;
  if (tile_n < 512) {                              // routed expert tile
    e = tile_n >> 5;                               // 32 col-tiles per expert
    const int i0 = (tile_n & 31) << 4;
    const bf16* base = wguT + ((size_t)e << 20);   // per-expert [2I][H]
    bg = base + (size_t)(i0 + l16) * H_DIM;        // gate rows [0,512)
    bu = base + (size_t)(I_DIM + i0 + l16) * H_DIM;// up   rows [512,1024)
    col0 = tile_n << 4;
  } else {                                         // shared expert tile
    const int i0 = (tile_n - 512) << 4;
    bg = sguT + (size_t)(i0 + l16) * H_DIM;
    bu = sguT + (size_t)(SI + i0 + l16) * H_DIM;
    col0 = E_NUM * I_DIM + i0;
  }
  const bf16* arow[4];
  arow[0] = xb + (size_t)(row0 + l16) * H_DIM;
  arow[1] = arow[0] + (size_t)16 * H_DIM;
  arow[2] = arow[0] + (size_t)32 * H_DIM;
  arow[3] = arow[0] + (size_t)48 * H_DIM;

  v8f cg[4] = {}, cu[4] = {};
  for (int k0 = 0; k0 < H_DIM; k0 += 32) {
    const int kA = k0 + half * 8;                  // 16-bit A/B lane layout
    V16U a[4], g, u;
    load_frag(g, bg + kA);
    load_frag(u, bu + kA);
#pragma unroll
    for (int mi = 0; mi < 4; ++mi) load_frag(a[mi], arow[mi] + kA);
#pragma unroll
    for (int mi = 0; mi < 4; ++mi) {
      cg[mi] = __builtin_amdgcn_wmma_f32_16x16x32_bf16(false, a[mi].v, false, g.v, (short)0, cg[mi], false, false);
      cu[mi] = __builtin_amdgcn_wmma_f32_16x16x32_bf16(false, a[mi].v, false, u.v, (short)0, cu[mi], false, false);
    }
  }
  const int n = col0 + l16;
#pragma unroll
  for (int mi = 0; mi < 4; ++mi) {
#pragma unroll
    for (int j = 0; j < 8; ++j) {                  // C tile: VGPR j -> M = j + half*8
      const int t = row0 + (mi << 4) + j + half * 8;
      const float wgt = (e >= 0) ? route[t * E_NUM + e] : 1.0f;
      const float gv = cg[mi][j], uv = cu[mi][j];
      const float sil = gv / (1.f + __expf(-gv));  // silu
      hbuf[(size_t)t * TOTI + n] = f2bf(sil * uv * wgt);
    }
  }
}

// ---------------- merged down-projection: out = hbuf[T,8704] @ Wdown_cat[8704,H] ----------------
// One wave owns a 32x64 output super-tile (2x4 16x16 tiles, 8 accumulators);
// 8 WMMAs per 6 fragment loads, 8 independent accumulation chains for ILP.
// 32 x 16 = 512 waves, 4 waves/block -> 128 blocks; K-loop of 272 steps.
__global__ void k_down(const bf16* __restrict__ hb,
                       const bf16* __restrict__ wdT,
                       float* __restrict__ out) {
  const int wave = threadIdx.x >> 5;
  const int lane = threadIdx.x & 31;
  const int half = lane >> 4;
  const int l16  = lane & 15;
  const int tile   = blockIdx.x * (blockDim.x >> 5) + wave;
  const int tile_n = tile & 15;                    // 0..15 (64 cols each)
  const int tile_m = tile >> 4;                    // 0..31 (32 rows each)
  const bf16* arow[2];
  const bf16* brow[4];
  arow[0] = hb + (size_t)((tile_m << 5) + l16) * TOTI;
  arow[1] = arow[0] + (size_t)16 * TOTI;
  brow[0] = wdT + (size_t)((tile_n << 6) + l16) * TOTI;
  brow[1] = brow[0] + (size_t)16 * TOTI;
  brow[2] = brow[0] + (size_t)32 * TOTI;
  brow[3] = brow[0] + (size_t)48 * TOTI;

  v8f c[2][4] = {};
  for (int k0 = 0; k0 < TOTI; k0 += 32) {
    const int kA = k0 + half * 8;
    V16U a[2], b[4];
    load_frag(a[0], arow[0] + kA);
    load_frag(a[1], arow[1] + kA);
#pragma unroll
    for (int ni = 0; ni < 4; ++ni) load_frag(b[ni], brow[ni] + kA);
    __builtin_prefetch(arow[0] + kA + 256, 0, 1);  // global_prefetch_b8 ahead in K
    __builtin_prefetch(brow[0] + kA + 256, 0, 1);
#pragma unroll
    for (int mi = 0; mi < 2; ++mi)
#pragma unroll
      for (int ni = 0; ni < 4; ++ni)
        c[mi][ni] = __builtin_amdgcn_wmma_f32_16x16x32_bf16(false, a[mi].v, false, b[ni].v,
                                                            (short)0, c[mi][ni], false, false);
  }
#pragma unroll
  for (int mi = 0; mi < 2; ++mi) {
#pragma unroll
    for (int ni = 0; ni < 4; ++ni) {
      const int n = (tile_n << 6) + (ni << 4) + l16;
#pragma unroll
      for (int j = 0; j < 8; ++j) {
        const int t = (tile_m << 5) + (mi << 4) + j + half * 8;
        out[(size_t)t * H_DIM + n] = c[mi][ni][j];
      }
    }
  }
}

// ---------------- host side ----------------

extern "C" void kernel_launch(void* const* d_in, const int* in_sizes, int n_in,
                              void* d_out, int out_size, void* d_ws, size_t ws_size,
                              hipStream_t stream) {
  (void)in_sizes; (void)n_in; (void)out_size; (void)ws_size;
  const float* x   = (const float*)d_in[0];   // [T,H]
  const float* gw  = (const float*)d_in[1];   // [E,H]
  const float* wgu = (const float*)d_in[2];   // [E,H,2I]
  const float* wd  = (const float*)d_in[3];   // [E,I,H]
  const float* sgu = (const float*)d_in[4];   // [H,2*SI]
  const float* sd  = (const float*)d_in[5];   // [SI,H]
  float* out = (float*)d_out;                 // [T,H] fp32

  char* ws = (char*)d_ws;
  size_t off = 0;
  auto alloc = [&](size_t bytes) { void* p = ws + off; off += (bytes + 255) & ~(size_t)255; return p; };
  unsigned short* xb   = (unsigned short*)alloc((size_t)T_TOK * H_DIM * 2);             //  2 MB
  unsigned short* wguT = (unsigned short*)alloc((size_t)E_NUM * 2 * I_DIM * H_DIM * 2); // 32 MB
  unsigned short* sguT = (unsigned short*)alloc((size_t)2 * SI * H_DIM * 2);            //  2 MB
  unsigned short* wdT  = (unsigned short*)alloc((size_t)H_DIM * TOTI * 2);              // 17.8 MB
  unsigned short* hb   = (unsigned short*)alloc((size_t)T_TOK * TOTI * 2);              // 17.8 MB
  float*          route = (float*)alloc((size_t)T_TOK * E_NUM * 4);                     // 64 KB

  k_cvt_x   <<<(T_TOK * H_DIM) / 256,              256, 0, stream>>>(x, xb);
  k_cvt_wguT<<<(E_NUM * 2 * I_DIM * H_DIM) / 256,  256, 0, stream>>>(wgu, wguT);
  k_cvt_sguT<<<(2 * SI * H_DIM) / 256,             256, 0, stream>>>(sgu, sguT);
  k_cvt_wdT <<<(H_DIM * TOTI) / 256,               256, 0, stream>>>(wd, sd, wdT);
  k_router  <<<T_TOK, 128, 0, stream>>>(x, gw, route);

  // 16 M-superblocks x 544 N-tiles, 8 waves (strips) per 256-thread block
  k_up_swiglu<<<(16 * 544) / 8, 256, 0, stream>>>((const bf16*)xb, (const bf16*)wguT,
                                                  (const bf16*)sguT, route, hb);
  // 32 x 16 super-tiles (32x64 each), 4 waves per 128-thread block
  k_down<<<512 / 4, 128, 0, stream>>>((const bf16*)hb, (const bf16*)wdT, out);
}
